// MetaPN_24988119728300
// MI455X (gfx1250) — matrix operand order repaired and checked by reading, not compiled, not run on hardware
//
#include <hip/hip_runtime.h>
#include <stdint.h>

typedef __bf16 bf16;
typedef __attribute__((ext_vector_type(16))) __bf16 v16bf;
typedef __attribute__((ext_vector_type(8)))  __bf16 v8bf;
typedef __attribute__((ext_vector_type(8)))  float  v8f;
typedef __attribute__((ext_vector_type(4)))  unsigned int u32x4;
typedef __attribute__((ext_vector_type(8)))  int i32x8;
typedef __attribute__((ext_vector_type(4)))  int i32x4;

#ifndef USE_TDM
#if __has_builtin(__builtin_amdgcn_tensor_load_to_lds) && __has_builtin(__builtin_amdgcn_s_wait_tensorcnt)
#define USE_TDM 1
#else
#define USE_TDM 0
#endif
#endif

#define HCOLS 1088   // concat cols: w1w(512) | bb1(256) | bb2(256) | bb3(64)

// ---------------- fragment loaders (LDS bf16, row stride 256) ----------------
// A-layout (16x32 bf16): lanes<16 hold K {ko..ko+7, ko+16..ko+23}, ko = kw*32 (+8 for hi lanes)
__device__ __forceinline__ v16bf load_fragA(const bf16* base, int row, int kw, int lane) {
    int ko = kw * 32 + ((lane & 16) ? 8 : 0);
    const bf16* p = base + row * 256 + ko;
    v8bf lo = *reinterpret_cast<const v8bf*>(p);
    v8bf hi = *reinterpret_cast<const v8bf*>(p + 16);
    v16bf f;
#pragma unroll
    for (int i = 0; i < 8; ++i) { f[i] = lo[i]; f[i + 8] = hi[i]; }
    return f;
}

// B-layout (32x16 bf16): lanes<16 hold K {ko..ko+7, ko+16..ko+23}, ko = kw*32 (+8 for hi lanes)
__device__ __forceinline__ v16bf load_fragB(const bf16* base, int col, int kw, int lane) {
    int ko = kw * 32 + ((lane & 16) ? 8 : 0);
    const bf16* p = base + col * 256 + ko;
    v8bf lo = *reinterpret_cast<const v8bf*>(p);
    v8bf hi = *reinterpret_cast<const v8bf*>(p + 16);
    v16bf f;
#pragma unroll
    for (int i = 0; i < 8; ++i) { f[i] = lo[i]; f[i + 8] = hi[i]; }
    return f;
}

// bf16-domain scale: legalizes to packed v_pk_mul_bf16 (VOP3P), co-executes with XDL WMMA
__device__ __forceinline__ v16bf scale_frag(v16bf f, bf16 s) {
    v16bf sv;
#pragma unroll
    for (int i = 0; i < 16; ++i) sv[i] = s;
    return f * sv;
}

// ---------------- TDM: flat 1-D bf16 tile global -> LDS ----------------
#if USE_TDM
__device__ __forceinline__ void tdm_load_flat(unsigned lds_off, const bf16* gsrc, unsigned nelem) {
    unsigned long long ga = (unsigned long long)(const void*)gsrc;
    u32x4 g0;
    g0[0] = 1u;                                           // count=1, user mode
    g0[1] = lds_off;                                      // lds_addr (bytes)
    g0[2] = (unsigned)ga;                                 // global_addr[31:0]
    g0[3] = (unsigned)((ga >> 32) & 0x1FFFFFFu) | (2u << 30); // addr[56:32] | type=2
    i32x8 g1;
    g1[0] = (int)(1u << 16);                              // wg_mask=0, data_size=1 (2B)
    g1[1] = (int)((nelem & 0xFFFFu) << 16);               // tensor_dim0 lo
    g1[2] = (int)(((nelem >> 16) & 0xFFFFu) | (1u << 16));// tensor_dim0 hi | tensor_dim1=1
    g1[3] = (int)((nelem & 0xFFFFu) << 16);               // tile_dim0
    g1[4] = 1;                                            // tile_dim1=1, tile_dim2=0
    g1[5] = (int)nelem;                                   // tensor_dim0_stride lo32
    g1[6] = (int)((nelem & 0xFFFFu) << 16);               // stride0 hi16=0 | stride1 lo16
    g1[7] = (int)(nelem >> 16);                           // stride1 rest
    i32x4 z4 = {0, 0, 0, 0};
    i32x8 z8 = {0, 0, 0, 0, 0, 0, 0, 0};
    // 6-arg form (clang-23 / therock-10.0 headers): groups 0..3, extra group, cpol
    __builtin_amdgcn_tensor_load_to_lds(g0, g1, z4, z4, z8, 0);
}
#endif

// ---------------- kernel 0: fp32 -> bf16 convert ----------------
__global__ void cvt_f32_bf16(const float* __restrict__ s, bf16* __restrict__ d, int n) {
    int i = blockIdx.x * blockDim.x + threadIdx.x;
    if (i < n) d[i] = (bf16)s[i];
}

// ---------------- kernel 1: H = pe @ [W1w|W1b|W2b|W3b]^T + bias ----------------
__global__ void __launch_bounds__(256)
hyper_gemm(const bf16* __restrict__ PEb,
           const bf16* __restrict__ W1w, const bf16* __restrict__ W1b,
           const bf16* __restrict__ W2b, const bf16* __restrict__ W3b,
           const float* __restrict__ b1w, const float* __restrict__ b1b,
           const float* __restrict__ b2b, const float* __restrict__ b3b,
           float* __restrict__ H) {
    const int tid = threadIdx.x, lane = tid & 31, wid = tid >> 5;
    const int wm = wid & 3, wn = wid >> 2;
    const int b0 = blockIdx.x * 128;
    const int ct = blockIdx.y;                    // 0..16 column tile over 1088 cols
    const bf16* Wseg; const float* bias;
    if (ct < 8)       { Wseg = W1w + (size_t)ct * 64 * 256;        bias = b1w + ct * 64; }
    else if (ct < 12) { Wseg = W1b + (size_t)(ct - 8) * 64 * 256;  bias = b1b + (ct - 8) * 64; }
    else if (ct < 16) { Wseg = W2b + (size_t)(ct - 12) * 64 * 256; bias = b2b + (ct - 12) * 64; }
    else              { Wseg = W3b;                                bias = b3b; }

    extern __shared__ char smem[];
    bf16* PEl = (bf16*)smem;              // [128][256]
    bf16* Ws  = PEl + 128 * 256;          // [64][256]
    {
        const uint4* sp = (const uint4*)(PEb + (size_t)b0 * 256);
        uint4* dp = (uint4*)PEl;
        for (int i = tid; i < 128 * 256 / 8; i += 256) dp[i] = sp[i];
        const uint4* sw = (const uint4*)Wseg;
        uint4* dw = (uint4*)Ws;
        for (int i = tid; i < 64 * 256 / 8; i += 256) dw[i] = sw[i];
    }
    __syncthreads();

    v8f zero;
#pragma unroll
    for (int i = 0; i < 8; ++i) zero[i] = 0.f;
    v8f acc[2][2];
#pragma unroll
    for (int mi = 0; mi < 2; ++mi)
#pragma unroll
        for (int ni = 0; ni < 2; ++ni) acc[mi][ni] = zero;

    int mrow[2] = { wm * 32 + (lane & 15), wm * 32 + 16 + (lane & 15) };
#pragma unroll
    for (int kw = 0; kw < 8; ++kw) {
        v16bf a[2], b[2];
#pragma unroll
        for (int mi = 0; mi < 2; ++mi) a[mi] = load_fragA(PEl, mrow[mi], kw, lane);
#pragma unroll
        for (int ni = 0; ni < 2; ++ni) b[ni] = load_fragB(Ws, wn * 32 + ni * 16 + (lane & 15), kw, lane);
#pragma unroll
        for (int mi = 0; mi < 2; ++mi)
#pragma unroll
            for (int ni = 0; ni < 2; ++ni)
                acc[mi][ni] = __builtin_amdgcn_wmma_f32_16x16x32_bf16(
                    false, a[mi], false, b[ni], (short)0, acc[mi][ni], false, false);
    }

#pragma unroll
    for (int mi = 0; mi < 2; ++mi)
#pragma unroll
        for (int ni = 0; ni < 2; ++ni) {
            int col = wn * 32 + ni * 16 + (lane & 15);
#pragma unroll
            for (int r = 0; r < 8; ++r) {
                int row = b0 + wm * 32 + mi * 16 + r + ((lane >> 4) << 3);
                H[(size_t)row * HCOLS + ct * 64 + col] = acc[mi][ni][r] + bias[col];
            }
        }
}

// ---------------- kernel 1b: x1 = PReLU(coods . w1 + bb1), store bf16 ----------------
__global__ void build_x1(const float* __restrict__ coods, const float* __restrict__ H,
                         const float* __restrict__ alpha_p, bf16* __restrict__ x1b) {
    int b = blockIdx.x, dd = threadIdx.x;
    float c0 = coods[b * 2 + 0], c1 = coods[b * 2 + 1];
    const float* h = H + (size_t)b * HCOLS;
    float v = c0 * h[dd] + c1 * h[256 + dd] + h[512 + dd];
    float alpha = alpha_p[0];
    v = (v >= 0.f) ? v : alpha * v;
    x1b[(size_t)b * 256 + dd] = (bf16)v;
}

// ---------------- kernels 2/3: fused hyper-layer mega-GEMM ----------------
// out[b,n] = PReLU?( sum_{d,k} X[b,d]*pe[b,k]*Wbig[(d,n),k] + sum_d X[b,d]*brs[d*Ntot+n]
//                    + H[b, Hoff+n] )
template<int MI, bool PRELU, bool OUT_BF16>
__global__ void __launch_bounds__(256)
meta_layer(const bf16* __restrict__ PEb, const bf16* __restrict__ Xb,
           const bf16* __restrict__ Wbig, const bf16* __restrict__ brs,
           const float* __restrict__ H, int Hoff,
           const float* __restrict__ alpha_p, int Ntot,
           bf16* __restrict__ outb, float* __restrict__ outf) {
    constexpr int BM = MI * 64;
    const int tid = threadIdx.x, lane = tid & 31, wid = tid >> 5;
    const int wm = wid & 3, wn = wid >> 2;
    const int b0 = blockIdx.x * BM;
    const int nbase = blockIdx.y * 64;

    extern __shared__ char smem[];
    bf16* PEl = (bf16*)smem;              // [BM][256] pe tile
    bf16* Xl  = PEl + BM * 256;           // [BM][256] activation tile
    bf16* Ws  = Xl + BM * 256;            // 2 x [64][256] W slabs (double buffer)

    {   // stage pe / x tiles (flat contiguous copies)
        const uint4* sp = (const uint4*)(PEb + (size_t)b0 * 256);
        const uint4* sx = (const uint4*)(Xb + (size_t)b0 * 256);
        uint4* dp = (uint4*)PEl;
        uint4* dx = (uint4*)Xl;
        for (int i = tid; i < BM * 256 / 8; i += 256) { dp[i] = sp[i]; dx[i] = sx[i]; }
    }
#if USE_TDM
    unsigned ws_offs[2];
    ws_offs[0] = (unsigned)(uintptr_t)(void*)Ws;
    ws_offs[1] = ws_offs[0] + 32768u;
    if (wid == 0) {   // prologue: slab for d=0
        tdm_load_flat(ws_offs[0], Wbig + (size_t)nbase * 256, 16384u);
        __builtin_amdgcn_s_wait_tensorcnt(0);
    }
#endif
    __syncthreads();

    int mrow[MI];
#pragma unroll
    for (int mi = 0; mi < MI; ++mi) mrow[mi] = wm * (MI * 16) + mi * 16 + (lane & 15);

    // preload pe A-fragments (reused for every d; only the x[b,d] scale changes)
    v16bf pef[MI][8];
#pragma unroll
    for (int mi = 0; mi < MI; ++mi)
#pragma unroll
        for (int kw = 0; kw < 8; ++kw) pef[mi][kw] = load_fragA(PEl, mrow[mi], kw, lane);

    v8f zero;
#pragma unroll
    for (int i = 0; i < 8; ++i) zero[i] = 0.f;
    v8f acc[MI][2];
#pragma unroll
    for (int mi = 0; mi < MI; ++mi)
#pragma unroll
        for (int ni = 0; ni < 2; ++ni) acc[mi][ni] = zero;

    for (int d = 0; d < 256; ++d) {
#if USE_TDM
        const bf16* Wcur = Ws + (d & 1) * 16384;
        if (wid == 0 && d + 1 < 256)
            tdm_load_flat(ws_offs[(d + 1) & 1],
                          Wbig + ((size_t)(d + 1) * Ntot + nbase) * 256, 16384u);
#else
        const bf16* Wcur = Ws;
        {
            const uint4* sw = (const uint4*)(Wbig + ((size_t)d * Ntot + nbase) * 256);
            uint4* dw = (uint4*)Ws;
            for (int i = tid; i < 16384 / 8; i += 256) dw[i] = sw[i];
        }
        __syncthreads();
#endif
        bf16 s[MI];
#pragma unroll
        for (int mi = 0; mi < MI; ++mi) s[mi] = Xl[mrow[mi] * 256 + d];

#pragma unroll
        for (int kw = 0; kw < 8; ++kw) {
            v16bf a[MI], b[2];
#pragma unroll
            for (int mi = 0; mi < MI; ++mi) a[mi] = scale_frag(pef[mi][kw], s[mi]);
#pragma unroll
            for (int ni = 0; ni < 2; ++ni)
                b[ni] = load_fragB(Wcur, wn * 32 + ni * 16 + (lane & 15), kw, lane);
#pragma unroll
            for (int mi = 0; mi < MI; ++mi)
#pragma unroll
                for (int ni = 0; ni < 2; ++ni)
                    acc[mi][ni] = __builtin_amdgcn_wmma_f32_16x16x32_bf16(
                        false, a[mi], false, b[ni], (short)0, acc[mi][ni], false, false);
        }
#if USE_TDM
        if (wid == 0) __builtin_amdgcn_s_wait_tensorcnt(0);
#endif
        __syncthreads();
    }

    // fold per-sample bias term: sum_d X[b,d] * brs[d*Ntot + n]  (8 extra K-steps)
    bf16* Wb = Ws;   // reuse slab buffer: LDS[n][k=d]
    for (int idx = tid; idx < 64 * 256; idx += 256) {
        int n = idx & 63, dd = idx >> 6;
        Wb[n * 256 + dd] = brs[(size_t)dd * Ntot + nbase + n];
    }
    __syncthreads();
#pragma unroll
    for (int kw = 0; kw < 8; ++kw) {
        v16bf a[MI], b[2];
#pragma unroll
        for (int mi = 0; mi < MI; ++mi) a[mi] = load_fragA(Xl, mrow[mi], kw, lane);
#pragma unroll
        for (int ni = 0; ni < 2; ++ni)
            b[ni] = load_fragB(Wb, wn * 32 + ni * 16 + (lane & 15), kw, lane);
#pragma unroll
        for (int mi = 0; mi < MI; ++mi)
#pragma unroll
            for (int ni = 0; ni < 2; ++ni)
                acc[mi][ni] = __builtin_amdgcn_wmma_f32_16x16x32_bf16(
                    false, a[mi], false, b[ni], (short)0, acc[mi][ni], false, false);
    }

    // epilogue: + bb (precomputed in H), PReLU, store
    const float alpha = alpha_p[0];
#pragma unroll
    for (int mi = 0; mi < MI; ++mi)
#pragma unroll
        for (int ni = 0; ni < 2; ++ni) {
            int col = nbase + wn * 32 + ni * 16 + (lane & 15);
#pragma unroll
            for (int r = 0; r < 8; ++r) {
                int row = b0 + wm * (MI * 16) + mi * 16 + r + ((lane >> 4) << 3);
                float v = acc[mi][ni][r] + H[(size_t)row * HCOLS + Hoff + col];
                if (PRELU) v = (v >= 0.f) ? v : alpha * v;
                if (OUT_BF16) outb[(size_t)row * Ntot + col] = (bf16)v;
                else          outf[(size_t)row * Ntot + col] = v;
            }
        }
}

// ---------------- host launcher ----------------
extern "C" void kernel_launch(void* const* d_in, const int* in_sizes, int n_in,
                              void* d_out, int out_size, void* d_ws, size_t ws_size,
                              hipStream_t stream) {
    const float* coods = (const float*)d_in[0];
    const float* pe    = (const float*)d_in[1];
    const float* W1w   = (const float*)d_in[2];
    const float* b1w   = (const float*)d_in[3];
    const float* W1b   = (const float*)d_in[4];
    const float* b1b   = (const float*)d_in[5];
    const float* W2w   = (const float*)d_in[6];
    const float* b2w   = (const float*)d_in[7];
    const float* W2b   = (const float*)d_in[8];
    const float* b2b   = (const float*)d_in[9];
    const float* W3w   = (const float*)d_in[10];
    const float* b3w   = (const float*)d_in[11];
    const float* W3b   = (const float*)d_in[12];
    const float* b3b   = (const float*)d_in[13];
    const float* alpha = (const float*)d_in[14];

    char* ws = (char*)d_ws;
    size_t off = 0;
    auto carve = [&](size_t bytes) -> void* {
        void* p = ws + off;
        off += (bytes + 255) & ~(size_t)255;
        return p;
    };
    bf16* W2w_bf = (bf16*)carve(16777216ull * 2);
    bf16* W3w_bf = (bf16*)carve(4194304ull * 2);
    bf16* W1w_bf = (bf16*)carve(131072ull * 2);
    bf16* W1b_bf = (bf16*)carve(65536ull * 2);
    bf16* W2b_bf = (bf16*)carve(65536ull * 2);
    bf16* W3b_bf = (bf16*)carve(16384ull * 2);
    bf16* b2w_bf = (bf16*)carve(65536ull * 2);
    bf16* b3w_bf = (bf16*)carve(16384ull * 2);
    bf16* pe_bf  = (bf16*)carve(1048576ull * 2);
    bf16* x1_bf  = (bf16*)carve(1048576ull * 2);
    bf16* x2_bf  = (bf16*)carve(1048576ull * 2);
    float* H     = (float*)carve(4456448ull * 4);   // [4096][1088]

    auto cvt = [&](const float* s, bf16* d, int n) {
        cvt_f32_bf16<<<(n + 255) / 256, 256, 0, stream>>>(s, d, n);
    };
    cvt(W2w, W2w_bf, 16777216);
    cvt(W3w, W3w_bf, 4194304);
    cvt(W1w, W1w_bf, 131072);
    cvt(W1b, W1b_bf, 65536);
    cvt(W2b, W2b_bf, 65536);
    cvt(W3b, W3b_bf, 16384);
    cvt(b2w, b2w_bf, 65536);
    cvt(b3w, b3w_bf, 16384);
    cvt(pe,  pe_bf,  1048576);

    // H = pe @ [W1w|W1b|W2b|W3b]^T + bias   (32 x 17 workgroups)
    hyper_gemm<<<dim3(32, 17), 256, 98304, stream>>>(
        pe_bf, W1w_bf, W1b_bf, W2b_bf, W3b_bf, b1w, b1b, b2b, b3b, H);

    build_x1<<<4096, 256, 0, stream>>>(coods, H, alpha, x1_bf);

    // layer 2: [4096,256] output, K=65536 fused GEMM
    meta_layer<2, true, true><<<dim3(32, 4), 256, 196608, stream>>>(
        pe_bf, x1_bf, W2w_bf, b2w_bf, H, 768, alpha, 256, x2_bf, nullptr);

    // layer 3: [4096,64] output -> d_out
    meta_layer<1, false, false><<<dim3(64, 1), 256, 131072, stream>>>(
        pe_bf, x2_bf, W3w_bf, b3w_bf, H, 1024, alpha, 64, nullptr, (float*)d_out);
}
